// DynamicVoxelPPEncoder_63591285785180
// MI455X (gfx1250) — matrix operand order repaired and checked by reading, not compiled, not run on hardware
//
#include <hip/hip_runtime.h>
#include <hip/hip_bf16.h>
#include <stdint.h>

typedef __attribute__((ext_vector_type(16))) _Float16 v16h;
typedef __attribute__((ext_vector_type(8)))  float    v8f;

#define NCELLS   262144
#define FIN      5
#define FOUT     64
#define BN_EPS   1e-3f

// Fragment table: each B-fragment is 32 lanes x 16 halves = 1024 B (512 halves)
#define FR_L1 0      // 4 frags  (K=32 pad from 5, N=64)
#define FR_L2 4      // 16 frags (K=64,  N=128) idx = s*8  + nt
#define FR_L3 20     // 64 frags (K=128, N=256) idx = s*16 + nt
#define FR_L4 84     // 32 frags (K=256, N=64)  idx = s*4  + nt
#define NFRAG 116
// bias offsets (floats): L1:0  L2:64  L3:192  L4:448  (total 512)

struct PP {
  const float *g0,*be0,*m0,*v0;
  const float *w1,*b1;
  const float *g1,*be1,*m1,*v1;
  const float *w2,*b2;
  const float *g2,*be2,*m2,*v2;
  const float *w3,*b3;
  const float *g3,*be3,*m3,*v3;
  const float *w4,*b4;
};

// ---------------- zero ----------------
__global__ void zero_kernel(float* __restrict__ p, int n) {
  int i = blockIdx.x * blockDim.x + threadIdx.x;
  if (i < n) p[i] = 0.0f;
}

// ---------------- prep: fold BN into f16 weight fragments + f32 biases ----------------
__global__ void prep_kernel(PP P, _Float16* __restrict__ wf, float* __restrict__ bias) {
  int t = blockIdx.x * blockDim.x + threadIdx.x;
  if (t < NFRAG * 32) {
    int f = t >> 5, lane = t & 31;
    int layer, s, nt;
    if (f < FR_L2)      { layer = 1; s = 0;                    nt = f; }
    else if (f < FR_L3) { int g = f - FR_L2; layer = 2; s = g >> 3; nt = g & 7; }
    else if (f < FR_L4) { int g = f - FR_L3; layer = 3; s = g >> 4; nt = g & 15; }
    else                { int g = f - FR_L4; layer = 4; s = g >> 2; nt = g & 3; }
    int n = nt * 16 + (lane & 15);
    int kbase = 32 * s + 16 * (lane >> 4);     // B layout: lanes 0-15 K=0-15, 16-31 K=16-31 (+32*s)
    for (int e = 0; e < 16; ++e) {
      int k = kbase + e;
      float w = 0.0f;
      if (layer == 1) {
        if (k < FIN) {
          float s0 = P.g0[k] * rsqrtf(P.v0[k] + BN_EPS);
          float s1 = P.g1[n] * rsqrtf(P.v1[n] + BN_EPS);
          w = P.w1[n * FIN + k] * s0 * s1;
        }
      } else if (layer == 2) {
        float s2 = P.g2[n] * rsqrtf(P.v2[n] + BN_EPS);
        w = P.w2[n * 64 + k] * s2;
      } else if (layer == 3) {
        float s3 = P.g3[n] * rsqrtf(P.v3[n] + BN_EPS);
        w = P.w3[n * 128 + k] * s3;
      } else {
        w = P.w4[n * 256 + k];
      }
      wf[f * 512 + lane * 16 + e] = (_Float16)w;
    }
  } else {
    int j = t - NFRAG * 32;
    if (j < 512) {
      if (j < 64) {
        int n = j;
        float s1 = P.g1[n] * rsqrtf(P.v1[n] + BN_EPS);
        float t1 = P.be1[n] - P.m1[n] * s1;
        float acc = P.b1[n];
        for (int k = 0; k < FIN; ++k) {
          float s0 = P.g0[k] * rsqrtf(P.v0[k] + BN_EPS);
          float t0 = P.be0[k] - P.m0[k] * s0;
          acc += t0 * P.w1[n * FIN + k];
        }
        bias[j] = s1 * acc + t1;
      } else if (j < 192) {
        int n = j - 64;
        float s2 = P.g2[n] * rsqrtf(P.v2[n] + BN_EPS);
        bias[j] = s2 * P.b2[n] + (P.be2[n] - P.m2[n] * s2);
      } else if (j < 448) {
        int n = j - 192;
        float s3 = P.g3[n] * rsqrtf(P.v3[n] + BN_EPS);
        bias[j] = s3 * P.b3[n] + (P.be3[n] - P.m3[n] * s3);
      } else {
        bias[j] = P.b4[j - 448];
      }
    }
  }
}

// Build an A-fragment (16x32 f16, contiguous k = 16*half + e) from per-wave LDS activation [16][W] f16.
// Lane: m = lane&15, half = lane>>4. VGPR t holds K pair (kb, kb+1).
__device__ inline v16h build_a(const _Float16* hb, int W, int s, int m, int half) {
  union { v16h v; uint32_t u[8]; } au;
#pragma unroll
  for (int t = 0; t < 8; ++t) {
    int kb = 2 * t + 16 * half + 32 * s;
    au.u[t] = *(const uint32_t*)(&hb[m * W + kb]);
  }
  return au.v;
}

#define WMMA(a, b, c) __builtin_amdgcn_wmma_f32_16x16x32_f16(false, (a), false, (b), (short)0, (c), false, false)

// ---------------- main: 1 wave = 16 points; MLP via WMMA; atomic scatter ----------------
__global__ void __launch_bounds__(128) pp_main(
    const float* __restrict__ points, float* __restrict__ out,
    float* __restrict__ counts, const _Float16* __restrict__ wf,
    const float* __restrict__ bias, int nTiles, int Nper, int BN)
{
  __shared__ _Float16 hb_all[4][16 * 256];   // 8 KB per wave
  __shared__ int cb_all[4][16];

  const int w    = threadIdx.x >> 5;
  const int lane = threadIdx.x & 31;
  const int half = lane >> 4;
  const int nl   = lane & 15;
  _Float16* hb = hb_all[w];
  int* cb = cb_all[w];

  const int tile = blockIdx.x * 4 + w;
  const bool valid = tile < nTiles;

  float pv0 = 0.f, pv1 = 0.f, pv2 = 0.f, pv3 = 0.f, pv4 = 0.f;
  bool vpoint = false;

  // ---- Stage 0/1: load points, compute cells, layer1 (K=32 pad, N=64) ----
  if (valid) {
    int raw = tile * 16 + nl;
    vpoint = raw < BN;
    int pIdx = vpoint ? raw : (BN - 1);
    const float* p = points + (size_t)pIdx * FIN;
    pv0 = p[0]; pv1 = p[1]; pv2 = p[2]; pv3 = p[3]; pv4 = p[4];
    if (half == 0) {
      int b  = pIdx / Nper;
      int ix = (int)((pv0 + 51.2f) * 5.0f);
      int iy = (int)((pv1 + 51.2f) * 5.0f);
      ix = min(max(ix, 0), 511);
      iy = min(max(iy, 0), 511);
      cb[nl] = vpoint ? (b * NCELLS + iy * 512 + ix) : -1;
    }

    v16h a1;
#pragma unroll
    for (int e = 0; e < 16; ++e) a1[e] = (_Float16)0.f;
    if (half == 0) {
      a1[0] = (_Float16)pv0; a1[1] = (_Float16)pv1; a1[2] = (_Float16)pv2;
      a1[3] = (_Float16)pv3; a1[4] = (_Float16)pv4;
    }
#pragma unroll
    for (int nt = 0; nt < 4; ++nt) {
      v16h bfr = *(const v16h*)(wf + (size_t)(FR_L1 + nt) * 512 + lane * 16);
      float bv = bias[nt * 16 + nl];
      v8f c;
#pragma unroll
      for (int j = 0; j < 8; ++j) c[j] = bv;
      c = WMMA(a1, bfr, c);
#pragma unroll
      for (int j = 0; j < 8; ++j)
        hb[(j + 8 * half) * 64 + nt * 16 + nl] = (_Float16)fmaxf(c[j], 0.f);
    }
  }
  __syncthreads();

  // ---- Layer 2: K=64 (2 steps), N=128 ----
  if (valid) {
    v16h a20 = build_a(hb, 64, 0, nl, half);
    v16h a21 = build_a(hb, 64, 1, nl, half);
#pragma unroll
    for (int nt = 0; nt < 8; ++nt) {
      v16h b0 = *(const v16h*)(wf + (size_t)(FR_L2 + 0 * 8 + nt) * 512 + lane * 16);
      v16h b1 = *(const v16h*)(wf + (size_t)(FR_L2 + 1 * 8 + nt) * 512 + lane * 16);
      float bv = bias[64 + nt * 16 + nl];
      v8f c;
#pragma unroll
      for (int j = 0; j < 8; ++j) c[j] = bv;
      c = WMMA(a20, b0, c);
      c = WMMA(a21, b1, c);
#pragma unroll
      for (int j = 0; j < 8; ++j)
        hb[(j + 8 * half) * 128 + nt * 16 + nl] = (_Float16)fmaxf(c[j], 0.f);
    }
  }
  __syncthreads();

  // ---- Layer 3: K=128 (4 steps), N=256 ----
  if (valid) {
    v16h a3[4];
#pragma unroll
    for (int s = 0; s < 4; ++s) a3[s] = build_a(hb, 128, s, nl, half);
    for (int nt = 0; nt < 16; ++nt) {
      float bv = bias[192 + nt * 16 + nl];
      v8f c;
#pragma unroll
      for (int j = 0; j < 8; ++j) c[j] = bv;
#pragma unroll
      for (int s = 0; s < 4; ++s) {
        v16h bfr = *(const v16h*)(wf + (size_t)(FR_L3 + s * 16 + nt) * 512 + lane * 16);
        c = WMMA(a3[s], bfr, c);
      }
#pragma unroll
      for (int j = 0; j < 8; ++j)
        hb[(j + 8 * half) * 256 + nt * 16 + nl] = (_Float16)fmaxf(c[j], 0.f);
    }
  }
  __syncthreads();

  // ---- Layer 4: K=256 (8 steps), N=64 + atomic scatter ----
  if (valid) {
    v16h a4[8];
#pragma unroll
    for (int s = 0; s < 8; ++s) a4[s] = build_a(hb, 256, s, nl, half);
    int cbm[8];
#pragma unroll
    for (int j = 0; j < 8; ++j) cbm[j] = cb[j + 8 * half];

#pragma unroll
    for (int nt = 0; nt < 4; ++nt) {
      float bv = bias[448 + nt * 16 + nl];
      v8f c;
#pragma unroll
      for (int j = 0; j < 8; ++j) c[j] = bv;
#pragma unroll
      for (int s = 0; s < 8; ++s) {
        v16h bfr = *(const v16h*)(wf + (size_t)(FR_L4 + s * 4 + nt) * 512 + lane * 16);
        c = WMMA(a4[s], bfr, c);
      }
#pragma unroll
      for (int j = 0; j < 8; ++j) {
        if (cbm[j] >= 0)
          atomicAdd(&out[(size_t)cbm[j] * 69 + 5 + nt * 16 + nl], c[j]);
      }
    }

    // raw 5 features + count
    if (half == 0 && vpoint) {
      int cell = cb[nl];
      size_t base = (size_t)cell * 69;
      atomicAdd(&out[base + 0], pv0);
      atomicAdd(&out[base + 1], pv1);
      atomicAdd(&out[base + 2], pv2);
      atomicAdd(&out[base + 3], pv3);
      atomicAdd(&out[base + 4], pv4);
      atomicAdd(&counts[cell], 1.0f);
    }
  }
}

// ---------------- finalize: mean ----------------
__global__ void fin_kernel(float* __restrict__ out, const float* __restrict__ counts, int total) {
  int i = blockIdx.x * blockDim.x + threadIdx.x;
  if (i < total) {
    int cell = i / 69;
    out[i] = out[i] / fmaxf(counts[cell], 1.0f);
  }
}

extern "C" void kernel_launch(void* const* d_in, const int* in_sizes, int n_in,
                              void* d_out, int out_size, void* d_ws, size_t ws_size,
                              hipStream_t stream) {
  const float* points = (const float*)d_in[0];
  // d_in[1] = base_index_mask (unused by the reference)

  // Detect param flattening order via sizes:
  //   insertion order: idx2 = bn0.gamma (size 5)
  //   pytree-sorted:   idx2 = b1 (size 64)
  const float* prm[24];
  if (n_in >= 26 && in_sizes[2] == FIN) {
    int order[24] = {2,3,4,5, 6,7, 8,9,10,11, 12,13, 14,15,16,17, 18,19, 20,21,22,23, 24,25};
    for (int i = 0; i < 24; ++i) prm[i] = (const float*)d_in[order[i]];
  } else {
    // sorted keys: b1 b2 b3 b4 bn0(4) bn1(4) bn2(4) bn3(4) w1 w2 w3 w4
    int order[24] = {6,7,8,9, 22,2, 10,11,12,13, 23,3, 14,15,16,17, 24,4, 18,19,20,21, 25,5};
    for (int i = 0; i < 24; ++i) prm[i] = (const float*)d_in[order[i]];
  }
  PP P;
  P.g0 = prm[0];  P.be0 = prm[1];  P.m0 = prm[2];  P.v0 = prm[3];
  P.w1 = prm[4];  P.b1  = prm[5];
  P.g1 = prm[6];  P.be1 = prm[7];  P.m1 = prm[8];  P.v1 = prm[9];
  P.w2 = prm[10]; P.b2  = prm[11];
  P.g2 = prm[12]; P.be2 = prm[13]; P.m2 = prm[14]; P.v2 = prm[15];
  P.w3 = prm[16]; P.b3  = prm[17];
  P.g3 = prm[18]; P.be3 = prm[19]; P.m3 = prm[20]; P.v3 = prm[21];
  P.w4 = prm[22]; P.b4  = prm[23];

  const int B    = out_size / (NCELLS * 69);     // = 2
  const int BN   = in_sizes[0] / FIN;            // total points = 500000
  const int Nper = BN / (B > 0 ? B : 1);         // points per batch
  const int nTiles = (BN + 15) / 16;

  // Workspace layout: [counts: B*NCELLS f32][weight frags: 116*512 f16][biases: 512 f32]
  float*    counts = (float*)d_ws;
  size_t    cbytes = (size_t)B * NCELLS * sizeof(float);
  _Float16* wf     = (_Float16*)((char*)d_ws + cbytes);
  float*    bias   = (float*)(wf + (size_t)NFRAG * 512);
  (void)ws_size;

  // 1) zero output + counts
  {
    int n = out_size;
    zero_kernel<<<(n + 255) / 256, 256, 0, stream>>>((float*)d_out, n);
    int nc = B * NCELLS;
    zero_kernel<<<(nc + 255) / 256, 256, 0, stream>>>(counts, nc);
  }
  // 2) fold BN into WMMA-layout weights/biases
  {
    int n = NFRAG * 32 + 512;
    prep_kernel<<<(n + 127) / 128, 128, 0, stream>>>(P, wf, bias);
  }
  // 3) main MLP + scatter
  {
    int blocks = (nTiles + 3) / 4;
    pp_main<<<blocks, 128, 0, stream>>>(points, (float*)d_out, counts, wf, bias,
                                        nTiles, Nper, BN);
  }
  // 4) mean
  {
    int n = out_size;
    fin_kernel<<<(n + 255) / 256, 256, 0, stream>>>((float*)d_out, counts, n);
  }
}
